// CliffordAttention_30442728194243
// MI455X (gfx1250) — matrix operand order repaired and checked by reading, not compiled, not run on hardware
//
#include <hip/hip_runtime.h>
#include <hip/hip_bf16.h>

// ---------------------------------------------------------------------------
// Clifford self-attention for MI455X (gfx1250), bf16 WMMA pipeline.
//   B=2, L=2048, H=8, head = D*NB = 128, DM = 1024.
// Stages:
//   1) convert query + weights fp32 -> bf16
//   2) proj GEMM (WMMA 16x16x32 bf16): Q (sign*scale folded), K, V (transposed)
//   3) fused logits -> softmax -> head-mean -> P@V per 16-row strip
// ---------------------------------------------------------------------------

#define BATCH 2
#define SEQ   2048
#define HEADS 8
#define HDIM  128          // D*NB
#define DMODEL 1024
#define ROWS  (BATCH * SEQ)    // 4096
#define LOGIT_SCALE 0.08838834764831845f   // 1/sqrt(128)

typedef __attribute__((ext_vector_type(16))) __bf16 v16bf;
typedef __attribute__((ext_vector_type(8)))  float  v8f;

union FragU { v16bf v; uint4 q[2]; };
union PackU { uint4 q; __bf16 e[8]; };

// ---- WMMA wrapper (codegen-confirmed builtin, 8 args) ----------------------
__device__ __forceinline__ v8f wmma_bf16(v16bf a, v16bf b, v8f c) {
  return __builtin_amdgcn_wmma_f32_16x16x32_bf16(
      /*neg_a=*/false, a, /*neg_b=*/false, b,
      /*c_mod=*/(short)0, c, /*reuse_a=*/false, /*reuse_b=*/false);
}

// A fragment: 16x32 bf16, lane = M row, K chunks (lane/16)*8 and +16.
__device__ __forceinline__ v16bf load_frag_a(const __bf16* base, int ld,
                                             int m0, int k0, int lane) {
  const int row = m0 + (lane & 15);
  const int kk  = k0 + ((lane >> 4) << 3);
  const __bf16* p = base + (size_t)row * ld + kk;
  FragU f;
  f.q[0] = *(const uint4*)(p);
  f.q[1] = *(const uint4*)(p + 16);
  return f.v;
}

// B fragment: 32x16 bf16, lane = N column; source row n0+lane%16, K chunks (lane/16)*8 and +16.
__device__ __forceinline__ v16bf load_frag_b(const __bf16* base, int ld,
                                             int n0, int k0, int lane) {
  const int row = n0 + (lane & 15);
  const int kk  = k0 + ((lane >> 4) << 3);
  const __bf16* p = base + (size_t)row * ld + kk;
  FragU f;
  f.q[0] = *(const uint4*)(p);
  f.q[1] = *(const uint4*)(p + 16);
  return f.v;
}

// ---- Kernel 1: fp32 -> bf16 ------------------------------------------------
__global__ void cvt_bf16_kernel(const float* __restrict__ src,
                                __bf16* __restrict__ dst, int n) {
  for (int i = blockIdx.x * blockDim.x + threadIdx.x; i < n;
       i += gridDim.x * blockDim.x)
    dst[i] = (__bf16)src[i];
}

// ---- Kernel 2: Y = X @ W^T + b (one 16x16 tile per wave) -------------------
// mode 0: Q  -> store bf16 row-major, fold blade sign * logit scale
// mode 1: K  -> store bf16 row-major
// mode 2: V  -> store bf16 transposed per (b,h): Vt[(b*H+h)*128 + c%128][l]
__global__ void proj_gemm_kernel(const __bf16* __restrict__ X,
                                 const __bf16* __restrict__ W,
                                 const float* __restrict__ bias,
                                 __bf16* __restrict__ Y,
                                 __bf16* __restrict__ Vt,
                                 int mode) {
  const int lane = threadIdx.x & 31;
  const int wv   = threadIdx.x >> 5;
  const int gw   = blockIdx.x * 8 + wv;           // global wave = tile id
  const int tn   = gw & 63;                       // DMODEL/16 = 64 col tiles
  const int tm   = gw >> 6;                       // ROWS/16 = 256 row tiles
  const int m0 = tm * 16, n0 = tn * 16;

  v8f acc = {};
#pragma unroll 4
  for (int kt = 0; kt < DMODEL / 32; ++kt) {
    v16bf a = load_frag_a(X, DMODEL, m0, kt * 32, lane);
    v16bf b = load_frag_b(W, DMODEL, n0, kt * 32, lane);   // W row n == (W^T) col n
    acc = wmma_bf16(a, b, acc);
  }

  const int c  = n0 + (lane & 15);                // output channel (N)
  const int rb = (lane >> 4) * 8;                 // first M row of this lane
  const float bv = bias[c];

  if (mode == 0) {                                // Q: reverse-sign + scale
    const float s = ((c & 7) < 4) ? LOGIT_SCALE : -LOGIT_SCALE;
#pragma unroll
    for (int v = 0; v < 8; ++v)
      Y[(size_t)(m0 + rb + v) * DMODEL + c] = (__bf16)((acc[v] + bv) * s);
  } else if (mode == 1) {                         // K
#pragma unroll
    for (int v = 0; v < 8; ++v)
      Y[(size_t)(m0 + rb + v) * DMODEL + c] = (__bf16)(acc[v] + bv);
  } else {                                        // V, transposed store
    const int b_  = m0 / SEQ;
    const int l0  = (m0 % SEQ) + rb;              // 8 consecutive rows
    const int h   = c / HDIM;
    const int cl  = c % HDIM;
    PackU p;
#pragma unroll
    for (int v = 0; v < 8; ++v) p.e[v] = (__bf16)(acc[v] + bv);
    __bf16* dst = Vt + ((size_t)(b_ * HEADS + h) * HDIM + cl) * SEQ + l0;
    *(uint4*)dst = p.q;
  }
}

// ---- Kernel 3: fused logits / softmax / head-mean / P@V --------------------
// One block = one 16-row strip of one batch; loops over all 8 heads.
__global__ void attn_kernel(const __bf16* __restrict__ Q,
                            const __bf16* __restrict__ K,
                            const __bf16* __restrict__ Vt,
                            float* __restrict__ out1,    // (B, L, DMODEL)
                            float* __restrict__ out2) {  // (B, L, L) head mean
  __shared__ __bf16 Plds[16 * SEQ];                      // 64 KB strip of P

  const int lane = threadIdx.x & 31;
  const int wv   = threadIdx.x >> 5;
  const int b_   = blockIdx.x / (SEQ / 16);
  const int l0   = (blockIdx.x % (SEQ / 16)) * 16;

  for (int h = 0; h < HEADS; ++h) {
    // ---------------- logits: S = Q_w K^T (scale pre-folded) ----------------
    const __bf16* Qb = Q + ((size_t)b_ * SEQ + l0) * DMODEL + h * HDIM;
    const __bf16* Kb = K + ((size_t)b_ * SEQ) * DMODEL + h * HDIM;
    v16bf afr[4];
#pragma unroll
    for (int kk = 0; kk < 4; ++kk)
      afr[kk] = load_frag_a(Qb, DMODEL, 0, kk * 32, lane);

    for (int t = wv; t < SEQ / 16; t += 8) {      // 16 column tiles per wave
      const int n0 = t * 16;
      v8f acc = {};
#pragma unroll
      for (int kk = 0; kk < 4; ++kk) {
        v16bf bf = load_frag_b(Kb, DMODEL, n0, kk * 32, lane);
        acc = wmma_bf16(afr[kk], bf, acc);
      }
      const int col = n0 + (lane & 15);
      const int rb  = (lane >> 4) * 8;
#pragma unroll
      for (int v = 0; v < 8; ++v)
        Plds[(rb + v) * SEQ + col] = (__bf16)acc[v];
    }
    __syncthreads();

    // ---------------- softmax: each wave owns 2 rows ------------------------
    for (int rr = 0; rr < 2; ++rr) {
      const int row = wv * 2 + rr;
      __bf16* rp = &Plds[row * SEQ];
      float mx = -3.4e38f;
      for (int i = lane; i < SEQ; i += 32) mx = fmaxf(mx, (float)rp[i]);
#pragma unroll
      for (int off = 16; off; off >>= 1) mx = fmaxf(mx, __shfl_xor(mx, off, 32));
      float sum = 0.f;
      for (int i = lane; i < SEQ; i += 32) {
        float e = __expf((float)rp[i] - mx);
        rp[i] = (__bf16)e;
        sum += e;
      }
#pragma unroll
      for (int off = 16; off; off >>= 1) sum += __shfl_xor(sum, off, 32);
      const float inv = 1.0f / sum;
      for (int i = lane; i < SEQ; i += 32)
        rp[i] = (__bf16)((float)rp[i] * inv);
    }
    __syncthreads();

    // ---------------- head mean (block exclusively owns rows: no atomics) ---
    {
      float* o2 = out2 + ((size_t)b_ * SEQ + l0) * SEQ;
      for (int e = threadIdx.x; e < 16 * SEQ; e += 256) {
        const float p = (float)Plds[e] * 0.125f;
        if (h == 0) o2[e] = p; else o2[e] += p;
      }
    }

    // ---------------- O = P @ V (one 16-channel tile per wave) --------------
    {
      const int n0 = wv * 16;
      const __bf16* Vb = Vt + (size_t)(b_ * HEADS + h) * HDIM * SEQ;
      v8f acc = {};
      for (int kk = 0; kk < SEQ / 32; ++kk) {
        v16bf a = load_frag_a((const __bf16*)Plds, SEQ, 0, kk * 32, lane);
        v16bf bf = load_frag_b(Vb, SEQ, n0, kk * 32, lane);
        acc = wmma_bf16(a, bf, acc);
      }
      const int c  = h * HDIM + n0 + (lane & 15);
      const int rb = (lane >> 4) * 8;
#pragma unroll
      for (int v = 0; v < 8; ++v)
        out1[((size_t)b_ * SEQ + l0 + rb + v) * DMODEL + c] = acc[v];
    }
    __syncthreads();   // protect Plds before next head's logits
  }
}

// ---------------------------------------------------------------------------
extern "C" void kernel_launch(void* const* d_in, const int* in_sizes, int n_in,
                              void* d_out, int out_size, void* d_ws, size_t ws_size,
                              hipStream_t stream) {
  const float* query = (const float*)d_in[0];
  const float* Wq    = (const float*)d_in[1];
  const float* bq    = (const float*)d_in[2];
  const float* Wk    = (const float*)d_in[3];
  const float* bk    = (const float*)d_in[4];
  const float* Wv    = (const float*)d_in[5];
  const float* bv    = (const float*)d_in[6];

  const size_t NX = (size_t)ROWS * DMODEL;        // 4,194,304
  const size_t NW = (size_t)DMODEL * DMODEL;      // 1,048,576

  __bf16* Xbf = (__bf16*)d_ws;
  __bf16* Wqb = Xbf + NX;
  __bf16* Wkb = Wqb + NW;
  __bf16* Wvb = Wkb + NW;
  __bf16* Qbf = Wvb + NW;
  __bf16* Kbf = Qbf + NX;
  __bf16* Vtb = Kbf + NX;                         // (B*H, 128, SEQ)

  float* out1 = (float*)d_out;                    // (B, L, DMODEL)
  float* out2 = out1 + NX;                        // (B, L, L)

  // Stage 1: bf16 conversion
  cvt_bf16_kernel<<<4096, 256, 0, stream>>>(query, Xbf, (int)NX);
  cvt_bf16_kernel<<<1024, 256, 0, stream>>>(Wq, Wqb, (int)NW);
  cvt_bf16_kernel<<<1024, 256, 0, stream>>>(Wk, Wkb, (int)NW);
  cvt_bf16_kernel<<<1024, 256, 0, stream>>>(Wv, Wvb, (int)NW);

  // Stage 2: projections (16384 tiles, 8 waves/block -> 2048 blocks)
  proj_gemm_kernel<<<2048, 256, 0, stream>>>(Xbf, Wqb, bq, Qbf, nullptr, 0);
  proj_gemm_kernel<<<2048, 256, 0, stream>>>(Xbf, Wkb, bk, Kbf, nullptr, 1);
  proj_gemm_kernel<<<2048, 256, 0, stream>>>(Xbf, Wvb, bv, nullptr, Vtb, 2);

  // Stage 3: fused attention, one block per 16-row strip per batch
  attn_kernel<<<BATCH * (SEQ / 16), 256, 0, stream>>>(Qbf, Kbf, Vtb, out1, out2);
}
